// nEMBlock_13297218748727
// MI455X (gfx1250) — matrix operand order repaired and checked by reading, not compiled, not run on hardware
//
#include <hip/hip_runtime.h>
#include <hip/hip_bf16.h>

typedef _Float16 v16h __attribute__((ext_vector_type(16)));
typedef float    v8f  __attribute__((ext_vector_type(8)));
typedef unsigned int u32x4 __attribute__((ext_vector_type(4)));
typedef int          i32x4 __attribute__((ext_vector_type(4)));
typedef int          i32x8 __attribute__((ext_vector_type(8)));

#define B_  4
#define L_  1024
#define D_  1024
#define NH_ 16
#define HD_ 64
#define M_  (B_ * L_)   // 4096 token rows

#if defined(__HIP_DEVICE_COMPILE__) && defined(__has_builtin)
#if __has_builtin(__builtin_amdgcn_tensor_load_to_lds) && \
    __has_builtin(__builtin_amdgcn_s_wait_tensorcnt)
#define HAVE_TDM 1
#endif
#endif
#ifndef HAVE_TDM
#define HAVE_TDM 0
#endif

// ---------------------------------------------------------------------------
// WMMA helpers (CDNA5: V_WMMA_F32_16X16X32_F16, wave32)
// ---------------------------------------------------------------------------
__device__ __forceinline__ v8f wmma_f16(v16h a, v16h b, v8f c) {
  return __builtin_amdgcn_wmma_f32_16x16x32_f16(false, a, false, b, (short)0, c,
                                                false, false);
}

// A-matrix 16x32 f16 layout (ISA 7.12.2): lane = m + 16*hf,
// elems 0..7 -> K = hf*8 + j, elems 8..15 -> K = 16 + hf*8 + j.
template <typename T>
__device__ __forceinline__ v16h load_a_tile(const T* Abase, int lda, int m,
                                            int hf, int k0) {
  const T* p0 = Abase + (size_t)m * lda + k0 + hf * 8;
  const T* p1 = p0 + 16;
  v16h r;
#pragma unroll
  for (int j = 0; j < 8; ++j) {
    r[j]     = (_Float16)p0[j];
    r[8 + j] = (_Float16)p1[j];
  }
  return r;
}

// B-matrix 32x16 f16 layout (same K order as A): elems 0..7 -> K = hf*8 + j,
// elems 8..15 -> K = 16 + hf*8 + j.
template <typename T>
__device__ __forceinline__ v16h load_b_tile(const T* row, int k0, int hf) {
  const T* p = row + k0 + hf * 8;
  v16h r;
#pragma unroll
  for (int j = 0; j < 8; ++j) {
    r[j]     = (_Float16)p[j];
    r[8 + j] = (_Float16)p[16 + j];
  }
  return r;
}

// ---------------------------------------------------------------------------
// Tensor Data Mover: async 2D tile (32 x rows64, f16) global -> LDS.
// D# built per CDNA5 ISA.
// ---------------------------------------------------------------------------
#if HAVE_TDM
__device__ __forceinline__ void tdm_load_b_tile(const _Float16* gsrc,
                                                _Float16* ldst, int td0,
                                                int stride, int rows_rem) {
  unsigned long long ga = (unsigned long long)(size_t)gsrc;
  unsigned lds = (unsigned)(size_t)ldst;  // flat->LDS: addr[31:0]
  u32x4 g0;
  g0.x = 1u;                                            // count=1 (valid D#)
  g0.y = lds;                                           // lds_addr
  g0.z = (unsigned)ga;                                  // global_addr[31:0]
  g0.w = (unsigned)((ga >> 32) & 0x01ffffffu) | (2u << 30);  // [56:32], type=2
  i32x8 g1;
  g1[0] = 1 << 16;                                      // data_size = 2B
  g1[1] = (td0 & 0xffff) << 16;                         // tensor_dim0[15:0]
  g1[2] = ((td0 >> 16) & 0xffff) | ((rows_rem & 0xffff) << 16);
  g1[3] = ((rows_rem >> 16) & 0xffff) | (32 << 16);     // tile_dim0 = 32
  g1[4] = 64;                                           // tile_dim1 = 64 rows
  g1[5] = stride;                                       // dim0_stride[31:0]
  g1[6] = 0;
  g1[7] = 0;
  i32x4 z4 = {0, 0, 0, 0};
#if __clang_major__ >= 23
  i32x8 z8 = {0, 0, 0, 0, 0, 0, 0, 0};
  __builtin_amdgcn_tensor_load_to_lds(g0, g1, z4, z4, z8, 0);
#else
  __builtin_amdgcn_tensor_load_to_lds(g0, g1, z4, z4, 0);
#endif
}
#else
// Fallback: cooperative staging (256 threads, 8 halfs each).
__device__ __forceinline__ void stage_copy(_Float16* dst, const _Float16* src,
                                           int ldsrc) {
  int idx = threadIdx.x * 8;
  int r = idx >> 5, c = idx & 31;
  const _Float16* s = src + (size_t)r * ldsrc + c;
#pragma unroll
  for (int j = 0; j < 8; ++j) dst[idx + j] = s[j];
}
#endif

// ---------------------------------------------------------------------------
// All-f16 GEMM:  C = A @ W^T.  A:[M,K] f16, W:[N,K] f16 row-major.
// Block = 256 threads = 8 waves; wave w -> rows m0=blockIdx.y*128+w*16,
// cols n0=blockIdx.x*64.  B tile (64x32) staged in LDS by the TDM,
// double-buffered; DMA of step k+1 overlaps WMMA on step k.
// FUSE: also compute paired vg tile (row offset vgRowOff) and emit
// act = (u*suv_u*32) * silu(vg*suv_v*32) in f16.
// ---------------------------------------------------------------------------
template <typename CT, bool FUSE>
__global__ __launch_bounds__(256) void gemm_f16k(
    const _Float16* __restrict__ A, const _Float16* __restrict__ W,
    CT* __restrict__ C, const float* __restrict__ suv, int N, int K,
    int vgRowOff) {
  __shared__ _Float16 lb[2][2][64 * 32];   // [stage][u/vg][tile]
  const int lane = threadIdx.x & 31;
  const int wv   = threadIdx.x >> 5;
  const int m    = lane & 15;
  const int hf   = lane >> 4;
  const int m0   = blockIdx.y * 128 + wv * 16;
  const int n0   = blockIdx.x * 64;

  v8f acc[4] = {}, accv[4] = {};
  const _Float16* Arow = A + (size_t)m0 * K;
  const _Float16* W0   = W + (size_t)n0 * K;
  const _Float16* W1   = W + (size_t)(vgRowOff + n0) * K;
  const int nk = K / 32;

  // Prologue: stage k-tile 0.
#if HAVE_TDM
  if (wv == 0) {
    tdm_load_b_tile(W0, lb[0][0], K, K, N - n0);
    if (FUSE) tdm_load_b_tile(W1, lb[0][1], K, K, N - n0);
  }
#else
  stage_copy(lb[0][0], W0, K);
  if (FUSE) stage_copy(lb[0][1], W1, K);
#endif

  for (int ik = 0; ik < nk; ++ik) {
    const int k0  = ik * 32;
    const int cur = ik & 1;
#if HAVE_TDM
    if (wv == 0) __builtin_amdgcn_s_wait_tensorcnt(0);
#endif
    __syncthreads();   // buffer `cur` ready for all waves
    if (ik + 1 < nk) {
      const int nxt = cur ^ 1;
      const int k1  = k0 + 32;
#if HAVE_TDM
      if (wv == 0) {
        tdm_load_b_tile(W0 + k1, lb[nxt][0], K - k1, K, N - n0);
        if (FUSE) tdm_load_b_tile(W1 + k1, lb[nxt][1], K - k1, K, N - n0);
      }
#else
      stage_copy(lb[nxt][0], W0 + k1, K);
      if (FUSE) stage_copy(lb[nxt][1], W1 + k1, K);
#endif
    }
    v16h a = load_a_tile(Arow, K, m, hf, k0);
    __builtin_prefetch(Arow + (size_t)m * K + k0 + 128, 0, 1);
#pragma unroll
    for (int t = 0; t < 4; ++t) {
      v16h b = load_b_tile(&lb[cur][0][(t * 16 + m) * 32], 0, hf);
      acc[t] = wmma_f16(a, b, acc[t]);
      if (FUSE) {
        v16h bg = load_b_tile(&lb[cur][1][(t * 16 + m) * 32], 0, hf);
        accv[t] = wmma_f16(a, bg, accv[t]);
      }
    }
    __syncthreads();   // all reads of `cur` done before it is re-filled
  }

#pragma unroll
  for (int t = 0; t < 4; ++t) {
    const int col = n0 + t * 16 + m;
#pragma unroll
    for (int v = 0; v < 8; ++v) {
      const int row = m0 + v + 8 * hf;
      const size_t idx = (size_t)row * N + col;
      if (FUSE) {
        float su = suv[col] * 32.0f;              // sqrt(D) = 32
        float sg = suv[vgRowOff + col] * 32.0f;
        float u  = acc[t][v] * su;
        float g  = accv[t][v] * sg;
        float si = g / (1.0f + __expf(-g));       // silu
        C[idx] = (CT)(u * si);
      } else {
        C[idx] = (CT)acc[t][v];
      }
    }
  }
}

// ---------------------------------------------------------------------------
// f32 -> f16 conversion (weights & activations), 4 elems/thread.
// ---------------------------------------------------------------------------
__global__ void cvt_f16_kernel(const float* __restrict__ in,
                               _Float16* __restrict__ out) {
  int i = (blockIdx.x * 256 + threadIdx.x) * 4;
#pragma unroll
  for (int j = 0; j < 4; ++j) out[i + j] = (_Float16)in[i + j];
}

// ---------------------------------------------------------------------------
// Cumulative sum of theta along L.
// ---------------------------------------------------------------------------
__global__ void cumsum_kernel(const float* __restrict__ theta,
                              float* __restrict__ thetaS) {
  int c  = blockIdx.x * blockDim.x + threadIdx.x;  // 0 .. B*512-1
  int b  = c >> 9;
  int ch = c & 511;
  const float* tp = theta  + (size_t)b * L_ * 512 + ch;
  float*       op = thetaS + (size_t)b * L_ * 512 + ch;
  float acc = 0.f;
  for (int l = 0; l < L_; ++l) {
    acc += tp[(size_t)l * 512];
    op[(size_t)l * 512] = acc;
  }
}

// ---------------------------------------------------------------------------
// Prep: normalized q,k (scaled), rotated positional qp,kp, v transposed to
// [b,h,hd,l].  One 32-lane wave per (b,l,h); lane t owns pair (2t, 2t+1).
// ---------------------------------------------------------------------------
__global__ __launch_bounds__(256) void prep_kernel(
    const float* __restrict__ qkv, const float* __restrict__ thetaS,
    const float* __restrict__ init_q, const float* __restrict__ init_k,
    const float* __restrict__ sqk_pi, const float* __restrict__ sqk,
    _Float16* __restrict__ qn, _Float16* __restrict__ kn,
    _Float16* __restrict__ qp, _Float16* __restrict__ kp,
    _Float16* __restrict__ vt) {
  const int t = threadIdx.x & 31;
  const int g = threadIdx.x >> 5;
  const int flat = blockIdx.x * 8 + g;       // b*L*NH + l*NH + h
  const int b   = flat >> 14;
  const int rem = flat & 16383;
  const int l   = rem >> 4;
  const int h   = rem & 15;

  float iqe = init_q[2 * t], iqo = init_q[2 * t + 1];
  float ike = init_k[2 * t], iko = init_k[2 * t + 1];
  float ssq = iqe * iqe + iqo * iqo;
  float ssk = ike * ike + iko * iko;
#pragma unroll
  for (int off = 16; off; off >>= 1) {
    ssq += __shfl_xor(ssq, off, 32);
    ssk += __shfl_xor(ssk, off, 32);
  }
  const float rq = rsqrtf(ssq), rk = rsqrtf(ssk);

  const float* qr = qkv + (size_t)(b * L_ + l) * (3 * D_);
  const int c0 = h * 64 + 2 * t, c1 = c0 + 1;
  float q0 = qr[c0],          q1 = qr[c1];
  float k0 = qr[D_ + c0],     k1 = qr[D_ + c1];
  float v0 = qr[2 * D_ + c0], v1 = qr[2 * D_ + c1];

  float sq2 = q0 * q0 + q1 * q1;
  float sk2 = k0 * k0 + k1 * k1;
#pragma unroll
  for (int off = 16; off; off >>= 1) {
    sq2 += __shfl_xor(sq2, off, 32);
    sk2 += __shfl_xor(sk2, off, 32);
  }
  const float rqn = rsqrtf(sq2), rkn = rsqrtf(sk2);
  const float s0 = sqk[c0] * 32.0f, s1 = sqk[c1] * 32.0f;   // 1/BASE = 32

  const size_t obase = ((size_t)(b * NH_ + h) * L_ + l) * 64;
  qn[obase + 2 * t]     = (_Float16)(q0 * rqn * s0);
  qn[obase + 2 * t + 1] = (_Float16)(q1 * rqn * s1);
  kn[obase + 2 * t]     = (_Float16)(k0 * rkn * s0);
  kn[obase + 2 * t + 1] = (_Float16)(k1 * rkn * s1);

  const size_t vbase = (size_t)(b * NH_ + h) * 64;
  vt[(vbase + 2 * t) * L_ + l]     = (_Float16)v0;
  vt[(vbase + 2 * t + 1) * L_ + l] = (_Float16)v1;

  const float th = thetaS[(size_t)(b * L_ + l) * 512 + h * 32 + t];
  float sn, cs;
  __sincosf(th, &sn, &cs);
  const float p0 = sqk_pi[c0] * 32.0f, p1 = sqk_pi[c1] * 32.0f;

  float ve = iqe * rq * p0, vo = iqo * rq * p1;
  qp[obase + 2 * t]     = (_Float16)(ve * cs - vo * sn);
  qp[obase + 2 * t + 1] = (_Float16)(ve * sn + vo * cs);
  ve = ike * rk * p0; vo = iko * rk * p1;
  kp[obase + 2 * t]     = (_Float16)(ve * cs - vo * sn);
  kp[obase + 2 * t + 1] = (_Float16)(ve * sn + vo * cs);
}

// ---------------------------------------------------------------------------
// Flash attention, logits = (q.k^T)(qp.kp^T)/HD^2, causal, online softmax,
// p@v — all WMMA.  One wave per 16 query rows, f16 output.
// ---------------------------------------------------------------------------
__global__ __launch_bounds__(256) void attn_kernel(
    const _Float16* __restrict__ qn, const _Float16* __restrict__ kn,
    const _Float16* __restrict__ qp, const _Float16* __restrict__ kp,
    const _Float16* __restrict__ vt, _Float16* __restrict__ y) {
  __shared__ _Float16 pl[8][16 * 32];
  const int lane = threadIdx.x & 31;
  const int wv   = threadIdx.x >> 5;
  const int m    = lane & 15;
  const int hf   = lane >> 4;
  const int bh = blockIdx.y;
  const int b = bh >> 4, h = bh & 15;
  const int q0 = blockIdx.x * 128 + wv * 16;

  const _Float16* qb  = qn + (size_t)bh * L_ * 64;
  const _Float16* kb  = kn + (size_t)bh * L_ * 64;
  const _Float16* qpb = qp + (size_t)bh * L_ * 64;
  const _Float16* kpb = kp + (size_t)bh * L_ * 64;
  const _Float16* vtb = vt + (size_t)bh * 64 * L_;   // [hd][L]

  v16h aq0 = load_a_tile(qb  + (size_t)q0 * 64, 64, m, hf, 0);
  v16h aq1 = load_a_tile(qb  + (size_t)q0 * 64, 64, m, hf, 32);
  v16h ap0 = load_a_tile(qpb + (size_t)q0 * 64, 64, m, hf, 0);
  v16h ap1 = load_a_tile(qpb + (size_t)q0 * 64, 64, m, hf, 32);

  v8f yacc[4] = {};
  float mrun[8], lrun[8];
#pragma unroll
  for (int v = 0; v < 8; ++v) { mrun[v] = -__builtin_inff(); lrun[v] = 0.f; }
  const float sc = 1.0f / (64.0f * 64.0f);   // (1/HD)^2
  _Float16* pw = &pl[wv][0];

  for (int kt = 0; kt < q0 + 16; kt += 32) {
    v8f s[2];
#pragma unroll
    for (int sub = 0; sub < 2; ++sub) {
      const _Float16* krow = kb + (size_t)(kt + sub * 16 + m) * 64;
      v16h bk0 = load_b_tile(krow, 0, hf);
      v16h bk1 = load_b_tile(krow, 32, hf);
      v8f ax = {};
      ax = wmma_f16(aq0, bk0, ax);
      ax = wmma_f16(aq1, bk1, ax);
      const _Float16* prow = kpb + (size_t)(kt + sub * 16 + m) * 64;
      v16h bp0 = load_b_tile(prow, 0, hf);
      v16h bp1 = load_b_tile(prow, 32, hf);
      v8f ag = {};
      ag = wmma_f16(ap0, bp0, ag);
      ag = wmma_f16(ap1, bp1, ag);
      v8f sv;
#pragma unroll
      for (int v = 0; v < 8; ++v) {
        float val = ax[v] * ag[v] * sc;
        int key = kt + sub * 16 + m;
        int row = q0 + v + 8 * hf;
        sv[v] = (key <= row) ? val : -__builtin_inff();
      }
      s[sub] = sv;
    }
#pragma unroll
    for (int v = 0; v < 8; ++v) {
      float mx = fmaxf(s[0][v], s[1][v]);
#pragma unroll
      for (int off = 8; off; off >>= 1) mx = fmaxf(mx, __shfl_xor(mx, off, 16));
      float mnew  = fmaxf(mrun[v], mx);
      float alpha = __expf(mrun[v] - mnew);
      float p0 = __expf(s[0][v] - mnew);
      float p1 = __expf(s[1][v] - mnew);
      float rs = p0 + p1;
#pragma unroll
      for (int off = 8; off; off >>= 1) rs += __shfl_xor(rs, off, 16);
      lrun[v] = lrun[v] * alpha + rs;
      mrun[v] = mnew;
#pragma unroll
      for (int t = 0; t < 4; ++t) yacc[t][v] *= alpha;
      int row = v + 8 * hf;
      pw[row * 32 + m]      = (_Float16)p0;
      pw[row * 32 + 16 + m] = (_Float16)p1;
    }
    v16h pa = load_a_tile(pw, 32, m, hf, 0);   // C-layout -> A-layout via LDS
#pragma unroll
    for (int t = 0; t < 4; ++t) {
      const _Float16* vrow = vtb + (size_t)(t * 16 + m) * L_ + kt;
      v16h bv = load_b_tile(vrow, 0, hf);
      yacc[t] = wmma_f16(pa, bv, yacc[t]);
    }
  }

#pragma unroll
  for (int t = 0; t < 4; ++t) {
#pragma unroll
    for (int v = 0; v < 8; ++v) {
      int row = q0 + v + 8 * hf;
      int col = h * 64 + t * 16 + m;
      y[((size_t)b * L_ + row) * D_ + col] = (_Float16)(yacc[t][v] / lrun[v]);
    }
  }
}

// ---------------------------------------------------------------------------
// Normalized residual: out = just_norm(An + |alpha*1.6|*(Bn - An)).
// Optionally mirrors the result to f16 (feeds next GEMM).
// ---------------------------------------------------------------------------
__device__ __forceinline__ float block_sum8(float v, float* red, int lane,
                                            int wv) {
#pragma unroll
  for (int off = 16; off; off >>= 1) v += __shfl_xor(v, off, 32);
  if (lane == 0) red[wv] = v;
  __syncthreads();
  float s = 0.f;
#pragma unroll
  for (int i = 0; i < 8; ++i) s += red[i];
  __syncthreads();
  return s;
}

__global__ __launch_bounds__(256) void resnorm_kernel(
    const float* __restrict__ xin, const float* __restrict__ hin,
    const float* __restrict__ alpha, float* __restrict__ out,
    _Float16* __restrict__ out_h) {
  __shared__ float red[8];
  const int lane = threadIdx.x & 31, wv = threadIdx.x >> 5;
  const size_t row = blockIdx.x;
  const float* xr = xin + row * D_;
  const float* hr = hin + row * D_;
  float xv[4], hv[4];
  float ssx = 0.f, ssh = 0.f;
#pragma unroll
  for (int i = 0; i < 4; ++i) {
    int e = threadIdx.x + i * 256;
    xv[i] = xr[e]; hv[i] = hr[e];
    ssx += xv[i] * xv[i];
    ssh += hv[i] * hv[i];
  }
  ssx = block_sum8(ssx, red, lane, wv);
  ssh = block_sum8(ssh, red, lane, wv);
  const float rx = rsqrtf(ssx), rh = rsqrtf(ssh);
  float tv[4], sst = 0.f;
#pragma unroll
  for (int i = 0; i < 4; ++i) {
    int e = threadIdx.x + i * 256;
    float an = xv[i] * rx, bn = hv[i] * rh;
    float lr = fabsf(alpha[e] * 1.6f);
    tv[i] = an + lr * (bn - an);
    sst += tv[i] * tv[i];
  }
  sst = block_sum8(sst, red, lane, wv);
  const float rt = rsqrtf(sst);
#pragma unroll
  for (int i = 0; i < 4; ++i) {
    int e = threadIdx.x + i * 256;
    float o = tv[i] * rt;
    out[row * D_ + e] = o;
    if (out_h) out_h[row * D_ + e] = (_Float16)o;
  }
}

// ---------------------------------------------------------------------------
// Host-side launch
// ---------------------------------------------------------------------------
extern "C" void kernel_launch(void* const* d_in, const int* in_sizes, int n_in,
                              void* d_out, int out_size, void* d_ws,
                              size_t ws_size, hipStream_t stream) {
  (void)in_sizes; (void)n_in; (void)out_size; (void)ws_size;
  const float* x      = (const float*)d_in[0];
  const float* Wqkv   = (const float*)d_in[1];
  const float* Watt   = (const float*)d_in[2];
  const float* Wfc    = (const float*)d_in[3];
  const float* Wmlp   = (const float*)d_in[4];
  const float* Wrot   = (const float*)d_in[5];
  const float* init_q = (const float*)d_in[6];
  const float* init_k = (const float*)d_in[7];
  const float* sqk_pi = (const float*)d_in[8];
  const float* sqk    = (const float*)d_in[9];
  const float* attn_a = (const float*)d_in[10];
  const float* mlp_a  = (const float*)d_in[11];
  const float* suv    = (const float*)d_in[12];

  float* out_final = (float*)d_out;
  float* theta_out = (float*)d_out + (size_t)M_ * D_;

  size_t off = 0;
  auto alloc = [&](size_t bytes) -> void* {
    void* p = (char*)d_ws + off;
    off += (bytes + 255) & ~(size_t)255;
    return p;
  };
  float*    qkv    = (float*)   alloc((size_t)M_ * 3072 * 4);
  float*    thetaS = (float*)   alloc((size_t)M_ * 512 * 4);
  _Float16* qn     = (_Float16*)alloc((size_t)B_ * NH_ * L_ * 64 * 2);
  _Float16* kn     = (_Float16*)alloc((size_t)B_ * NH_ * L_ * 64 * 2);
  _Float16* qp     = (_Float16*)alloc((size_t)B_ * NH_ * L_ * 64 * 2);
  _Float16* kp     = (_Float16*)alloc((size_t)B_ * NH_ * L_ * 64 * 2);
  _Float16* vt     = (_Float16*)alloc((size_t)B_ * NH_ * 64 * L_ * 2);
  _Float16* yh     = (_Float16*)alloc((size_t)M_ * D_ * 2);
  float*    h_att  = (float*)   alloc((size_t)M_ * D_ * 4);
  float*    x1     = (float*)   alloc((size_t)M_ * D_ * 4);
  _Float16* x1h    = (_Float16*)alloc((size_t)M_ * D_ * 2);
  _Float16* act    = (_Float16*)alloc((size_t)M_ * 4096 * 2);
  float*    h_mlp  = (float*)   alloc((size_t)M_ * D_ * 4);
  _Float16* xh     = (_Float16*)alloc((size_t)M_ * D_ * 2);
  _Float16* Wqkv_h = (_Float16*)alloc((size_t)3072 * D_ * 2);
  _Float16* Watt_h = (_Float16*)alloc((size_t)D_ * D_ * 2);
  _Float16* Wfc_h  = (_Float16*)alloc((size_t)8192 * D_ * 2);
  _Float16* Wmlp_h = (_Float16*)alloc((size_t)D_ * 4096 * 2);
  _Float16* Wrot_h = (_Float16*)alloc((size_t)512 * D_ * 2);

  const dim3 blk(256);
  // 0) one-time f32 -> f16 of GEMM operands
  cvt_f16_kernel<<<dim3((M_ * D_) / 1024), blk, 0, stream>>>(x, xh);
  cvt_f16_kernel<<<dim3((3072 * D_) / 1024), blk, 0, stream>>>(Wqkv, Wqkv_h);
  cvt_f16_kernel<<<dim3((D_ * D_) / 1024), blk, 0, stream>>>(Watt, Watt_h);
  cvt_f16_kernel<<<dim3((8192 * D_) / 1024), blk, 0, stream>>>(Wfc, Wfc_h);
  cvt_f16_kernel<<<dim3((D_ * 4096) / 1024), blk, 0, stream>>>(Wmlp, Wmlp_h);
  cvt_f16_kernel<<<dim3((512 * D_) / 1024), blk, 0, stream>>>(Wrot, Wrot_h);

  // 1) qkv = x @ Wqkv^T
  gemm_f16k<float, false><<<dim3(3072 / 64, M_ / 128), blk, 0, stream>>>(
      xh, Wqkv_h, qkv, nullptr, 3072, D_, 0);
  // 2) theta = x @ Wrot^T  -> second output slot
  gemm_f16k<float, false><<<dim3(512 / 64, M_ / 128), blk, 0, stream>>>(
      xh, Wrot_h, theta_out, nullptr, 512, D_, 0);
  // 3) thetaS = cumsum_L(theta)
  cumsum_kernel<<<dim3((B_ * 512) / 256), blk, 0, stream>>>(theta_out, thetaS);
  // 4) prep q/k/qp/kp/v (f16)
  prep_kernel<<<dim3(B_ * L_ * NH_ / 8), blk, 0, stream>>>(
      qkv, thetaS, init_q, init_k, sqk_pi, sqk, qn, kn, qp, kp, vt);
  // 5) flash attention -> y (f16)
  attn_kernel<<<dim3(L_ / 128, B_ * NH_), blk, 0, stream>>>(qn, kn, qp, kp, vt,
                                                            yh);
  // 6) h_att = y @ Watt^T
  gemm_f16k<float, false><<<dim3(D_ / 64, M_ / 128), blk, 0, stream>>>(
      yh, Watt_h, h_att, nullptr, D_, D_, 0);
  // 7) x1 = resnorm(x, h_att, attn_alpha)  (+ f16 mirror)
  resnorm_kernel<<<dim3(M_), blk, 0, stream>>>(x, h_att, attn_a, x1, x1h);
  // 8) act = u * silu(vg), fused into Wfc GEMM, f16 out
  gemm_f16k<_Float16, true><<<dim3(4096 / 64, M_ / 128), blk, 0, stream>>>(
      x1h, Wfc_h, act, suv, 4096, D_, 4096);
  // 9) h_mlp = act @ Wmlp^T   (K = 4096)
  gemm_f16k<float, false><<<dim3(D_ / 64, M_ / 128), blk, 0, stream>>>(
      act, Wmlp_h, h_mlp, nullptr, D_, 4096, 0);
  // 10) out = resnorm(x1, h_mlp, mlp_alpha)
  resnorm_kernel<<<dim3(M_), blk, 0, stream>>>(x1, h_mlp, mlp_a, out_final,
                                               nullptr);
}
